// HoloLayer_74448963109045
// MI455X (gfx1250) — matrix operand order repaired and checked by reading, not compiled, not run on hardware
//
#include <hip/hip_runtime.h>
#include <hip/hip_bf16.h>
#include <math.h>

typedef __attribute__((ext_vector_type(16))) __bf16       bf16x16;
typedef __attribute__((ext_vector_type(8)))  float        f32x8;
typedef __attribute__((ext_vector_type(4)))  unsigned int u32x4;
typedef __attribute__((ext_vector_type(4)))  int          i32x4v;

#define D_MODEL 1024
#define SIXD    (6*D_MODEL)
#define NHEAD   8
#define DHEAD   128
#define FFDIM   4096
#define BATCH   4
#define SEQ     4096
#define ROWS    (BATCH*SEQ)      // 16384
#define EPSV    1e-6f
#define NCHUNK  8
#define CHLEN   (SEQ/NCHUNK)     // 512

// ---------------------------------------------------------------------------
// CDNA5 async global->LDS copy (ASYNCcnt-tracked), guarded; sync fallback.
// ---------------------------------------------------------------------------
#if defined(__has_builtin)
# if __has_builtin(__builtin_amdgcn_global_load_async_to_lds_b128) && \
     __has_builtin(__builtin_amdgcn_s_wait_asynccnt)
#  define ASYNC_LDS 1
# endif
#endif

__device__ __forceinline__ void copy16(__bf16* l, const __bf16* g) {
#if defined(ASYNC_LDS)
  __builtin_amdgcn_global_load_async_to_lds_b128(
      (__attribute__((address_space(1))) i32x4v*)g,
      (__attribute__((address_space(3))) i32x4v*)l, 0, 0);
#else
  *(u32x4*)l = *(const u32x4*)g;
#endif
}
__device__ __forceinline__ void wait_async() {
#if defined(ASYNC_LDS)
  __builtin_amdgcn_s_wait_asynccnt(0);
#endif
}

// ---------------------------------------------------------------------------
// mod = silu(cond) @ W_adaln + b_adaln     [4, 6144]
// ---------------------------------------------------------------------------
__global__ void mod_kernel(const float* __restrict__ cond,
                           const float* __restrict__ W,
                           const float* __restrict__ bias,
                           float* __restrict__ mod) {
  int g   = blockIdx.x * blockDim.x + threadIdx.x;   // 0 .. 4*6144-1
  int b   = g / SIXD;
  int col = g % SIXD;
  const float* c = cond + (size_t)b * D_MODEL;
  float acc = 0.f;
  for (int k = 0; k < D_MODEL; ++k) {
    float cv = c[k];
    float s  = cv / (1.f + __expf(-cv));
    acc += s * W[(size_t)k * SIXD + col];
  }
  mod[(size_t)b * SIXD + col] = acc + bias[col];
}

// ---------------------------------------------------------------------------
// f32 [K][N]  ->  bf16 [N][K]   (transpose + downconvert, 32x32 LDS tiles)
// ---------------------------------------------------------------------------
__global__ void transpose_bf16_kernel(const float* __restrict__ W,
                                      __bf16* __restrict__ Wt,
                                      int K, int N) {
  __shared__ float tile[32][33];
  int tx = threadIdx.x, ty = threadIdx.y;
  int k0 = blockIdx.y * 32, n0 = blockIdx.x * 32;
  for (int i = 0; i < 4; ++i) {
    int kr = ty + i * 8;
    tile[kr][tx] = W[(size_t)(k0 + kr) * N + n0 + tx];
  }
  __syncthreads();
  for (int i = 0; i < 4; ++i) {
    int nr = ty + i * 8;
    Wt[(size_t)(n0 + nr) * K + k0 + tx] = (__bf16)tile[tx][nr];
  }
}

// ---------------------------------------------------------------------------
// AdaLN: LayerNorm(no affine) then x*(1+scale)+shift  -> bf16 rows
// scale_seg: 0 for (s_a,sh_a), 3 for (s_m,sh_m)
// ---------------------------------------------------------------------------
__global__ void adaln_kernel(const float* __restrict__ x,
                             const float* __restrict__ mod,
                             int scale_seg,
                             __bf16* __restrict__ out) {
  int row = blockIdx.x;
  int b   = row >> 12;                       // 4096 rows per batch
  const float* xr = x + (size_t)row * D_MODEL;
  const float* sc = mod + (size_t)b * SIXD + (size_t)scale_seg * D_MODEL;
  const float* sh = sc + D_MODEL;
  int tid = threadIdx.x;
  __shared__ float red[256];

  float v[4];
  float sum = 0.f;
  for (int i = 0; i < 4; ++i) { v[i] = xr[tid + i * 256]; sum += v[i]; }
  red[tid] = sum; __syncthreads();
  for (int off = 128; off > 0; off >>= 1) {
    if (tid < off) red[tid] += red[tid + off];
    __syncthreads();
  }
  float mu = red[0] * (1.f / 1024.f);
  __syncthreads();
  float s2 = 0.f;
  for (int i = 0; i < 4; ++i) { float d = v[i] - mu; s2 += d * d; }
  red[tid] = s2; __syncthreads();
  for (int off = 128; off > 0; off >>= 1) {
    if (tid < off) red[tid] += red[tid + off];
    __syncthreads();
  }
  float rstd = rsqrtf(red[0] * (1.f / 1024.f) + EPSV);
  for (int i = 0; i < 4; ++i) {
    int col  = tid + i * 256;
    float xn = (v[i] - mu) * rstd;
    float o  = xn * (1.f + sc[col]) + sh[col];
    out[(size_t)row * D_MODEL + col] = (__bf16)o;
  }
}

// ---------------------------------------------------------------------------
// bf16 WMMA GEMM: C[M,N] = A[M,K] @ Bt[N,K]^T, f32 accum.
// Block tile 256x128, 8 waves (4 M x 2 N), wave tile 64x64 -> 16 WMMA / K-step.
// Double-buffered LDS; async global->LDS when available.
// EPI: 0 = store f32 ; 1 = exact-GELU -> bf16 ; 2 = resid + gate*acc -> f32
// ---------------------------------------------------------------------------
union Frag { bf16x16 v; u32x4 q[2]; };

template<int EPI>
__global__ __launch_bounds__(256) void gemm_wmma_kernel(
    const __bf16* __restrict__ A, const __bf16* __restrict__ B,
    int M, int N, int K,
    float* __restrict__ Cf, __bf16* __restrict__ Cb,
    const float* __restrict__ resid, const float* __restrict__ mod,
    int gate_off) {
  constexpr int LDT = 40;                    // 80B rows: 16B aligned, bank-spread
  __shared__ __bf16 As[2][256 * LDT];
  __shared__ __bf16 Bs[2][128 * LDT];

  int tid  = threadIdx.x;
  int lane = tid & 31, wid = tid >> 5;
  int wm = wid >> 1, wn = wid & 1;           // 4 x 2 wave grid
  int m0 = blockIdx.y * 256, n0 = blockIdx.x * 128;
  int l16 = lane & 15, lhalf = lane >> 4;

  f32x8 zero = {0.f, 0.f, 0.f, 0.f, 0.f, 0.f, 0.f, 0.f};
  f32x8 acc[4][4];
  for (int mt = 0; mt < 4; ++mt)
    for (int nt = 0; nt < 4; ++nt) acc[mt][nt] = zero;

  auto stage = [&](int k0, int buf) {
    for (int i = 0; i < 4; ++i) {            // A: 256x32 = 1024 16B chunks
      int c = tid + i * 256;
      int row = c >> 2, kc = (c & 3) << 3;
      copy16(&As[buf][row * LDT + kc], A + (size_t)(m0 + row) * K + k0 + kc);
    }
    for (int i = 0; i < 2; ++i) {            // B: 128x32 = 512 16B chunks
      int c = tid + i * 256;
      int row = c >> 2, kc = (c & 3) << 3;
      copy16(&Bs[buf][row * LDT + kc], B + (size_t)(n0 + row) * K + k0 + kc);
    }
  };

  stage(0, 0);
  wait_async();
  __syncthreads();

  int cur = 0;
  for (int k0 = 0; k0 < K; k0 += 32) {
    if (k0 + 32 < K) stage(k0 + 32, cur ^ 1);   // prefetch next tile

    // A 16x32 bf16 fragment: lanes0-15 K{0-7,16-23}, lanes16-31 K{8-15,24-31}
    Frag a[4];
    for (int mt = 0; mt < 4; ++mt) {
      const __bf16* p = &As[cur][(wm * 64 + mt * 16 + l16) * LDT + lhalf * 8];
      a[mt].q[0] = *(const u32x4*)p;
      a[mt].q[1] = *(const u32x4*)(p + 16);
    }
    // B 32x16 bf16 fragment: lanes0-15 K{0-7,16-23}, lanes16-31 K{8-15,24-31}
    for (int nt = 0; nt < 4; ++nt) {
      Frag bf;
      const __bf16* p = &Bs[cur][(wn * 64 + nt * 16 + l16) * LDT + lhalf * 8];
      bf.q[0] = *(const u32x4*)p;
      bf.q[1] = *(const u32x4*)(p + 16);
      for (int mt = 0; mt < 4; ++mt)
        acc[mt][nt] = __builtin_amdgcn_wmma_f32_16x16x32_bf16(
            false, a[mt].v, false, bf.v, (short)0, acc[mt][nt],
            false, false);
    }
    wait_async();
    __syncthreads();
    cur ^= 1;
  }

  // epilogue: C layout — VGPR r, lanes0-15 -> M=r, lanes16-31 -> M=8+r, N=lane%16
  for (int mt = 0; mt < 4; ++mt)
    for (int nt = 0; nt < 4; ++nt) {
      int col = n0 + wn * 64 + nt * 16 + l16;
      for (int r = 0; r < 8; ++r) {
        int row   = m0 + wm * 64 + mt * 16 + r + lhalf * 8;
        float v   = acc[mt][nt][r];
        size_t ix = (size_t)row * N + col;
        if (EPI == 0) {
          Cf[ix] = v;
        } else if (EPI == 1) {
          float g = 0.5f * v * (1.f + erff(v * 0.70710678118f));
          Cb[ix] = (__bf16)g;
        } else {
          int b      = row >> 12;
          float gate = mod[(size_t)b * SIXD + gate_off + col];
          Cf[ix] = resid[ix] + gate * v;
        }
      }
    }
}

// ---------------------------------------------------------------------------
// HRR bind: kv[i] = sum_j k[j] * v[(i-j) mod 128]   (one block per token-head)
// ---------------------------------------------------------------------------
__global__ void hrr_conv_kernel(const float* __restrict__ qkv,
                                float* __restrict__ kv) {
  int bid = blockIdx.x;            // (b*S+s)*H + h
  int h   = bid & 7;
  int bs  = bid >> 3;
  int tid = threadIdx.x;           // 0..127
  __shared__ float kk[DHEAD], vv[DHEAD];
  size_t base = (size_t)bs * 3 * D_MODEL + h * DHEAD;
  kk[tid] = qkv[base + D_MODEL + tid];
  vv[tid] = qkv[base + 2 * D_MODEL + tid];
  __syncthreads();
  float acc = 0.f;
  for (int j = 0; j < DHEAD; ++j)
    acc += kk[j] * vv[(tid - j) & (DHEAD - 1)];
  kv[(size_t)bs * D_MODEL + h * DHEAD + tid] = acc;
}

// ---------------------------------------------------------------------------
// chunked causal cumsum over S (in-place) + per-chunk totals
// ---------------------------------------------------------------------------
__global__ void cumsum_chunk_kernel(float* __restrict__ kv,
                                    float* __restrict__ totals) {
  int bid  = blockIdx.x;           // BATCH * NCHUNK * 4 = 128 blocks
  int dgrp = bid & 3;
  int ch   = (bid >> 2) & 7;
  int b    = bid >> 5;
  int d    = dgrp * 256 + threadIdx.x;
  float run = 0.f;
  size_t base = ((size_t)b * SEQ + (size_t)ch * CHLEN) * D_MODEL + d;
  for (int s = 0; s < CHLEN; ++s) {
    size_t ix = base + (size_t)s * D_MODEL;
    run += kv[ix];
    kv[ix] = run;
  }
  totals[((size_t)b * NCHUNK + ch) * D_MODEL + d] = run;
}

__global__ void scan_totals_kernel(float* __restrict__ totals) {
  int bid  = blockIdx.x;           // BATCH * 4
  int dgrp = bid & 3;
  int b    = bid >> 2;
  int d    = dgrp * 256 + threadIdx.x;
  float run = 0.f;
  for (int ch = 0; ch < NCHUNK; ++ch) {
    size_t ix = ((size_t)b * NCHUNK + ch) * D_MODEL + d;
    float t = totals[ix];
    totals[ix] = run;              // exclusive chunk offset
    run += t;
  }
}

// ---------------------------------------------------------------------------
// HRR unbind: out[i] = sum_j q[j] * kvc[(i+j) mod 128], kvc = cumsum + offset
// ---------------------------------------------------------------------------
__global__ void hrr_corr_kernel(const float* __restrict__ qkv,
                                const float* __restrict__ kv,
                                const float* __restrict__ totals,
                                __bf16* __restrict__ attn) {
  int bid = blockIdx.x;
  int h   = bid & 7;
  int bs  = bid >> 3;
  int s   = bs & (SEQ - 1);
  int b   = bs >> 12;
  int ch  = s / CHLEN;
  int tid = threadIdx.x;
  __shared__ float kc[DHEAD], qq[DHEAD];
  int d = h * DHEAD + tid;
  kc[tid] = kv[(size_t)bs * D_MODEL + d] +
            totals[((size_t)b * NCHUNK + ch) * D_MODEL + d];
  qq[tid] = qkv[(size_t)bs * 3 * D_MODEL + d];
  __syncthreads();
  float acc = 0.f;
  for (int j = 0; j < DHEAD; ++j)
    acc += qq[j] * kc[(tid + j) & (DHEAD - 1)];
  attn[(size_t)bs * D_MODEL + d] = (__bf16)acc;
}

// ---------------------------------------------------------------------------
extern "C" void kernel_launch(void* const* d_in, const int* in_sizes, int n_in,
                              void* d_out, int out_size, void* d_ws,
                              size_t ws_size, hipStream_t stream) {
  const float* x       = (const float*)d_in[0];
  const float* cond    = (const float*)d_in[1];
  const float* W_adaln = (const float*)d_in[2];
  const float* b_adaln = (const float*)d_in[3];
  const float* W_qkv   = (const float*)d_in[4];
  const float* W_o     = (const float*)d_in[5];
  const float* W_fc    = (const float*)d_in[6];
  const float* W_out   = (const float*)d_in[7];
  float* out = (float*)d_out;

  char* p = (char*)d_ws;
  size_t off = 0;
  auto alloc = [&](size_t bytes) -> char* {
    char* r = p + off;
    off += (bytes + 255) & ~(size_t)255;
    return r;
  };
  float*  mod    = (float*) alloc((size_t)BATCH * SIXD * 4);
  __bf16* wqkvt  = (__bf16*)alloc((size_t)3072 * 1024 * 2);
  __bf16* wot    = (__bf16*)alloc((size_t)1024 * 1024 * 2);
  __bf16* wfct   = (__bf16*)alloc((size_t)4096 * 1024 * 2);
  __bf16* woutt  = (__bf16*)alloc((size_t)1024 * 4096 * 2);
  __bf16* xn     = (__bf16*)alloc((size_t)ROWS * D_MODEL * 2);
  float*  qkv    = (float*) alloc((size_t)ROWS * 3 * D_MODEL * 4);
  float*  kv     = (float*) alloc((size_t)ROWS * D_MODEL * 4);
  float*  totals = (float*) alloc((size_t)BATCH * NCHUNK * D_MODEL * 4);
  __bf16* attn   = (__bf16*)alloc((size_t)ROWS * D_MODEL * 2);
  float*  x2     = (float*) alloc((size_t)ROWS * D_MODEL * 4);
  __bf16* xn2    = (__bf16*)alloc((size_t)ROWS * D_MODEL * 2);
  __bf16* h      = (__bf16*)alloc((size_t)ROWS * FFDIM * 2);

  // 1. modulation vector
  mod_kernel<<<(BATCH * SIXD) / 256, 256, 0, stream>>>(cond, W_adaln, b_adaln, mod);

  // 2. weight transpose + bf16
  transpose_bf16_kernel<<<dim3(3072 / 32, 1024 / 32), dim3(32, 8), 0, stream>>>(W_qkv, wqkvt, 1024, 3072);
  transpose_bf16_kernel<<<dim3(1024 / 32, 1024 / 32), dim3(32, 8), 0, stream>>>(W_o, wot, 1024, 1024);
  transpose_bf16_kernel<<<dim3(4096 / 32, 1024 / 32), dim3(32, 8), 0, stream>>>(W_fc, wfct, 1024, 4096);
  transpose_bf16_kernel<<<dim3(1024 / 32, 4096 / 32), dim3(32, 8), 0, stream>>>(W_out, woutt, 4096, 1024);

  // 3. adaLN (attention branch)
  adaln_kernel<<<ROWS, 256, 0, stream>>>(x, mod, 0, xn);

  // 4. qkv = xn @ W_qkv
  gemm_wmma_kernel<0><<<dim3(3072 / 128, ROWS / 256), 256, 0, stream>>>(
      xn, wqkvt, ROWS, 3072, 1024, qkv, nullptr, nullptr, nullptr, 0);

  // 5. bind, 6-7. causal cumsum, 8. unbind
  hrr_conv_kernel<<<ROWS * NHEAD, DHEAD, 0, stream>>>(qkv, kv);
  cumsum_chunk_kernel<<<BATCH * NCHUNK * 4, 256, 0, stream>>>(kv, totals);
  scan_totals_kernel<<<BATCH * 4, 256, 0, stream>>>(totals);
  hrr_corr_kernel<<<ROWS * NHEAD, DHEAD, 0, stream>>>(qkv, kv, totals, attn);

  // 9. x2 = x + g_a * (attn @ W_o)
  gemm_wmma_kernel<2><<<dim3(1024 / 128, ROWS / 256), 256, 0, stream>>>(
      attn, wot, ROWS, 1024, 1024, x2, nullptr, x, mod, 2 * D_MODEL);

  // 10. adaLN (mlp branch)
  adaln_kernel<<<ROWS, 256, 0, stream>>>(x2, mod, 3, xn2);

  // 11. h = gelu(xn2 @ W_fc)
  gemm_wmma_kernel<1><<<dim3(4096 / 128, ROWS / 256), 256, 0, stream>>>(
      xn2, wfct, ROWS, 4096, 1024, nullptr, h, nullptr, nullptr, 0);

  // 12. out = x2 + g_m * (h @ W_out)
  gemm_wmma_kernel<2><<<dim3(1024 / 128, ROWS / 256), 256, 0, stream>>>(
      h, woutt, ROWS, 1024, 4096, out, nullptr, x2, mod, 5 * D_MODEL);
}
